// QuantumNeuralNetwork_50964081934952
// MI455X (gfx1250) — matrix operand order repaired and checked by reading, compiled, not run on hardware
//
#include <hip/hip_runtime.h>
#include <hip/hip_bf16.h>

typedef __attribute__((ext_vector_type(16))) __bf16 v16bf;
typedef __attribute__((ext_vector_type(8)))  float  v8f;

#define BM 256
#define BN 256
#define BK 32
#define LDK 36      // u32 row stride: 144B, 16B-aligned
#define NTHREADS 512

// Split f32 into two bf16 halves packed in one u32: low16 = hi(bf16), high16 = lo(bf16).
__device__ __forceinline__ unsigned pack_bf16x2(float a) {
    unsigned ua    = __float_as_uint(a);
    unsigned hbits = ua & 0xffff0000u;
    float    r     = a - __uint_as_float(hbits);
    unsigned lbits = __float_as_uint(r) & 0xffff0000u;
    return (hbits >> 16) | lbits;
}

__device__ __forceinline__ void unpack_pair(unsigned p0, unsigned p1,
                                            unsigned& h, unsigned& l) {
    h = (p0 & 0xffffu) | (p1 << 16);
    l = (p0 >> 16)     | (p1 & 0xffff0000u);
}

union Frag16 { unsigned u[8]; v16bf v; };

// A fragment (16x32, ISA A-layout): lane m = lane&15, kb = 8*(lane>>4).
__device__ __forceinline__ void load_fragA(const unsigned* srow, int kb,
                                           v16bf& hi, v16bf& lo) {
    uint4 g0 = *(const uint4*)(srow + kb);
    uint4 g1 = *(const uint4*)(srow + kb + 4);
    uint4 g2 = *(const uint4*)(srow + kb + 16);
    uint4 g3 = *(const uint4*)(srow + kb + 20);
    Frag16 H, L;
    unpack_pair(g0.x, g0.y, H.u[0], L.u[0]);
    unpack_pair(g0.z, g0.w, H.u[1], L.u[1]);
    unpack_pair(g1.x, g1.y, H.u[2], L.u[2]);
    unpack_pair(g1.z, g1.w, H.u[3], L.u[3]);
    unpack_pair(g2.x, g2.y, H.u[4], L.u[4]);
    unpack_pair(g2.z, g2.w, H.u[5], L.u[5]);
    unpack_pair(g3.x, g3.y, H.u[6], L.u[6]);
    unpack_pair(g3.z, g3.w, H.u[7], L.u[7]);
    hi = H.v; lo = L.v;
}

// B fragment (32x16): lane n = lane&15, kb = 8*(lane>>4); K = kb..kb+7, kb+16..kb+23.
__device__ __forceinline__ void load_fragB(const unsigned* srow,
                                           v16bf& hi, v16bf& lo) {
    uint4 g0 = *(const uint4*)(srow);
    uint4 g1 = *(const uint4*)(srow + 4);
    uint4 g2 = *(const uint4*)(srow + 16);
    uint4 g3 = *(const uint4*)(srow + 20);
    Frag16 H, L;
    unpack_pair(g0.x, g0.y, H.u[0], L.u[0]);
    unpack_pair(g0.z, g0.w, H.u[1], L.u[1]);
    unpack_pair(g1.x, g1.y, H.u[2], L.u[2]);
    unpack_pair(g1.z, g1.w, H.u[3], L.u[3]);
    unpack_pair(g2.x, g2.y, H.u[4], L.u[4]);
    unpack_pair(g2.z, g2.w, H.u[5], L.u[5]);
    unpack_pair(g3.x, g3.y, H.u[6], L.u[6]);
    unpack_pair(g3.z, g3.w, H.u[7], L.u[7]);
    hi = H.v; lo = L.v;
}

// Issue async copy of one 256x32-u32 tile (rows of 128B) into LDS (ASYNCcnt-tracked).
// 4 wave-instructions per wave; per lane 16B.
__device__ __forceinline__ void issue_async_tile(const unsigned* gbase, int ldke,
                                                 int row0, int k0,
                                                 unsigned lds_base, int tid) {
    #pragma unroll
    for (int i = 0; i < 4; ++i) {
        int task = tid + i * NTHREADS;       // 0..2047
        int r    = task >> 3;                // row 0..255
        int seg  = task & 7;                 // 16B segment 0..7
        unsigned lds_off = lds_base + (unsigned)(r * (LDK * 4) + seg * 16);
        const unsigned* g = gbase + (size_t)(row0 + r) * ldke + k0 + seg * 4;
        asm volatile("global_load_async_to_lds_b128 %0, %1, off"
                     :: "v"(lds_off), "v"(g) : "memory");
    }
}

// quantum_probs: batch-independent, depth fixed = 3.
__global__ void qprobs_kernel(const float* __restrict__ qw, float* __restrict__ qp, int H) {
    int h = blockIdx.x * blockDim.x + threadIdx.x;
    if (h >= H) return;
    float s0r = 1.f, s0i = 0.f, s1r = 0.f, s1i = 0.f;
    #pragma unroll
    for (int d = 0; d < 3; ++d) {
        float th = qw[(size_t)(d * H + h) * 4 + 0];
        float ph = qw[(size_t)(d * H + h) * 4 + 3];
        float c  = __cosf(th), s  = __sinf(th);
        float n0r = c * s0r - s * s1r, n0i = c * s0i - s * s1i;
        float n1r = s * s0r + c * s1r, n1i = s * s0i + c * s1i;
        float cp = __cosf(ph), sp = __sinf(ph);
        s0r = n0r; s0i = n0i;
        s1r = n1r * cp - n1i * sp;
        s1i = n1r * sp + n1i * cp;
    }
    qp[h] = s0r * s0r + s0i * s0i;
}

// Elementwise pack: f32 -> packed bf16x2 u32.
__global__ void packx_kernel(const float* __restrict__ in, unsigned* __restrict__ out, long n) {
    long i = (long)blockIdx.x * blockDim.x + threadIdx.x;
    if (i < n) out[i] = pack_bf16x2(in[i]);
}

// Transpose + pack: W[K][N] f32 -> Wt[N][K] packed u32. 32x32 LDS tiles, 32x8 threads.
__global__ void packT_kernel(const float* __restrict__ W, unsigned* __restrict__ Wt,
                             int K, int N) {
    __shared__ float t[32][33];
    int n0 = blockIdx.x * 32, k0 = blockIdx.y * 32;
    int tx = threadIdx.x, ty = threadIdx.y;
    #pragma unroll
    for (int j = 0; j < 32; j += 8)
        t[ty + j][tx] = W[(size_t)(k0 + ty + j) * N + n0 + tx];
    __syncthreads();
    #pragma unroll
    for (int j = 0; j < 32; j += 8)
        Wt[(size_t)(n0 + ty + j) * K + k0 + tx] = pack_bf16x2(t[tx][ty + j]);
}

// C = act(A @ B + bias (+qp)). A: packed u32 [M][K]. Bt: packed u32 [N][K].
// 256x256x32 tiles, 512 threads = 16 waves (4x4), 64x64 per wave.
// Double-buffered LDS filled by GLOBAL_LOAD_ASYNC_TO_LDS_B128; ONE barrier per k-step:
//   wait(own tile t) -> barrier -> issue(t+1 into other buf) -> compute(tile t).
// The barrier proves all waves finished iteration t-1's reads of the buffer t+1 targets.
// OUTP: 1 = write packed u32 (feeds next layer), 0 = write f32.
template <int ACT, int OUTP>
__global__ __launch_bounds__(NTHREADS)
void gemm_async_kernel(const unsigned* __restrict__ AP, const unsigned* __restrict__ BtP,
                       const float* __restrict__ bias, const float* __restrict__ qp,
                       void* __restrict__ Cv, int M, int N, int K) {
    __shared__ unsigned sA[2][BM * LDK];
    __shared__ unsigned sB[2][BN * LDK];

    // L2-friendly grid swizzle (1D launch)
    const int grid_m = M >> 8;
    const int grid_n = N >> 8;
    const int GROUP  = 8;
    int pid    = blockIdx.x;
    int nig    = GROUP * grid_n;
    int group  = pid / nig;
    int firstm = group * GROUP;
    int gsz    = (grid_m - firstm < GROUP) ? (grid_m - firstm) : GROUP;
    int bm     = firstm + (pid % gsz);
    int bn     = (pid % nig) / gsz;
    const int m0 = bm << 8;
    const int n0 = bn << 8;

    const int tid  = threadIdx.x;
    const int lane = tid & 31;
    const int wid  = tid >> 5;
    const int wm   = wid >> 2;
    const int wn   = wid & 3;
    const int lrow = lane & 15;
    const int kbA  = (lane >> 4) << 3;
    const int kbB  = (lane >> 4) << 3;

    // Raw LDS byte offsets (low 32 bits of the generic pointer = LDS offset).
    unsigned offA[2], offB[2];
    offA[0] = (unsigned)(uintptr_t)&sA[0][0];
    offA[1] = (unsigned)(uintptr_t)&sA[1][0];
    offB[0] = (unsigned)(uintptr_t)&sB[0][0];
    offB[1] = (unsigned)(uintptr_t)&sB[1][0];

    v8f acc[4][4];
    #pragma unroll
    for (int mt = 0; mt < 4; ++mt)
        #pragma unroll
        for (int nt = 0; nt < 4; ++nt)
            acc[mt][nt] = (v8f){0.f, 0.f, 0.f, 0.f, 0.f, 0.f, 0.f, 0.f};

    const int nsteps = K / BK;
    // Prologue: tile 0 -> buffer 0
    issue_async_tile(AP,  K, m0, 0, offA[0], tid);
    issue_async_tile(BtP, K, n0, 0, offB[0], tid);

    for (int t = 0; t < nsteps; ++t) {
        const int cur = t & 1;
        // Tile t is the only set of async ops this wave has outstanding.
        asm volatile("s_wait_asynccnt 0x0" ::: "memory");
        __syncthreads();  // everyone has tile t; everyone finished t-1's reads

        if (t + 1 < nsteps) {
            const int k1 = (t + 1) * BK;
            issue_async_tile(AP,  K, m0, k1, offA[cur ^ 1], tid);
            issue_async_tile(BtP, K, n0, k1, offB[cur ^ 1], tid);
        }

        v16bf bh[4], bl[4];
        #pragma unroll
        for (int nt = 0; nt < 4; ++nt) {
            const unsigned* srow = &sB[cur][0] + (size_t)(wn * 64 + nt * 16 + lrow) * LDK + kbB;
            load_fragB(srow, bh[nt], bl[nt]);
        }
        #pragma unroll
        for (int mt = 0; mt < 4; ++mt) {
            v16bf ah, al;
            const unsigned* srow = &sA[cur][0] + (size_t)(wm * 64 + mt * 16 + lrow) * LDK;
            load_fragA(srow, kbA, ah, al);
            #pragma unroll
            for (int nt = 0; nt < 4; ++nt) {
                acc[mt][nt] = __builtin_amdgcn_wmma_f32_16x16x32_bf16(
                    false, ah, false, bh[nt], (short)0, acc[mt][nt], false, false);
                acc[mt][nt] = __builtin_amdgcn_wmma_f32_16x16x32_bf16(
                    false, ah, false, bl[nt], (short)0, acc[mt][nt], false, false);
                acc[mt][nt] = __builtin_amdgcn_wmma_f32_16x16x32_bf16(
                    false, al, false, bh[nt], (short)0, acc[mt][nt], false, false);
            }
        }
    }

    // Epilogue: bias (+qprobs) + activation
    float*    Cf = (float*)Cv;
    unsigned* Cu = (unsigned*)Cv;
    const int rbase = (lane >> 4) << 3;
    #pragma unroll
    for (int mt = 0; mt < 4; ++mt) {
        #pragma unroll
        for (int nt = 0; nt < 4; ++nt) {
            int col = n0 + wn * 64 + nt * 16 + lrow;
            float badd = bias[col] + (qp ? qp[col] : 0.0f);
            #pragma unroll
            for (int r = 0; r < 8; ++r) {
                int row = m0 + wm * 64 + mt * 16 + rbase + r;
                float v = acc[mt][nt][r] + badd;
                v = (ACT == 0) ? fmaxf(v, 0.0f) : tanhf(v);
                size_t idx = (size_t)row * N + col;
                if (OUTP) Cu[idx] = pack_bf16x2(v);
                else      Cf[idx] = v;
            }
        }
    }
}

extern "C" void kernel_launch(void* const* d_in, const int* in_sizes, int n_in,
                              void* d_out, int out_size, void* d_ws, size_t ws_size,
                              hipStream_t stream) {
    const float* x     = (const float*)d_in[0];
    const float* w0    = (const float*)d_in[1];
    const float* b0    = (const float*)d_in[2];
    const float* w1    = (const float*)d_in[3];
    const float* b1    = (const float*)d_in[4];
    const float* q1_qw = (const float*)d_in[5];
    const float* q1_cw = (const float*)d_in[6];
    const float* q1_cb = (const float*)d_in[7];
    const float* w2    = (const float*)d_in[8];
    const float* b2    = (const float*)d_in[9];
    const float* q2_qw = (const float*)d_in[10];
    const float* q2_cw = (const float*)d_in[11];
    const float* q2_cb = (const float*)d_in[12];
    const float* w3    = (const float*)d_in[13];
    const float* b3    = (const float*)d_in[14];
    float* out = (float*)d_out;

    const int H     = in_sizes[2];
    const int D_IN  = in_sizes[1] / H;
    const int Bsz   = in_sizes[0] / D_IN;
    const int D_OUT = in_sizes[14];

    float* wsf = (float*)d_ws;
    float*    qp1 = wsf;
    float*    qp2 = wsf + H;
    unsigned* xP  = (unsigned*)(wsf + 2 * H);
    unsigned* hA  = xP + (size_t)Bsz * D_IN;
    unsigned* hB  = hA + (size_t)Bsz * H;
    unsigned* wt  = hB + (size_t)Bsz * H;   // H*H u32 repack buffer (reused)

    qprobs_kernel<<<(H + 255) / 256, 256, 0, stream>>>(q1_qw, qp1, H);
    qprobs_kernel<<<(H + 255) / 256, 256, 0, stream>>>(q2_qw, qp2, H);

    long nx = (long)Bsz * D_IN;
    packx_kernel<<<(unsigned)((nx + 255) / 256), 256, 0, stream>>>(x, xP, nx);

    dim3 blk(NTHREADS);
    dim3 gH((Bsz / BM) * (H / BN));
    dim3 gO((Bsz / BM) * (D_OUT / BN));
    dim3 tb(32, 8);
    dim3 tgH(H / 32, H / 32);          // [K=H][N=H]
    dim3 tg0(H / 32, D_IN / 32);       // w0: [K=D_IN][N=H]
    dim3 tg3(D_OUT / 32, H / 32);      // w3: [K=H][N=D_OUT]

    packT_kernel<<<tg0, tb, 0, stream>>>(w0, wt, D_IN, H);
    gemm_async_kernel<0, 1><<<gH, blk, 0, stream>>>(xP, wt, b0, nullptr, hA, Bsz, H, D_IN);

    packT_kernel<<<tgH, tb, 0, stream>>>(w1, wt, H, H);
    gemm_async_kernel<0, 1><<<gH, blk, 0, stream>>>(hA, wt, b1, nullptr, hB, Bsz, H, H);

    packT_kernel<<<tgH, tb, 0, stream>>>(q1_cw, wt, H, H);
    gemm_async_kernel<1, 1><<<gH, blk, 0, stream>>>(hB, wt, q1_cb, qp1, hA, Bsz, H, H);

    packT_kernel<<<tgH, tb, 0, stream>>>(w2, wt, H, H);
    gemm_async_kernel<0, 1><<<gH, blk, 0, stream>>>(hA, wt, b2, nullptr, hB, Bsz, H, H);

    packT_kernel<<<tgH, tb, 0, stream>>>(q2_cw, wt, H, H);
    gemm_async_kernel<1, 1><<<gH, blk, 0, stream>>>(hB, wt, q2_cb, qp2, hA, Bsz, H, H);

    packT_kernel<<<tg3, tb, 0, stream>>>(w3, wt, H, D_OUT);
    gemm_async_kernel<0, 0><<<gO, blk, 0, stream>>>(hA, wt, b3, nullptr, out, Bsz, D_OUT, H);
}
